// VlbComponents_61108794687744
// MI455X (gfx1250) — matrix operand order repaired and checked by reading, not compiled, not run on hardware
//
#include <hip/hip_runtime.h>
#include <hip/hip_bf16.h>

// ---------------------------------------------------------------------------
// CDNA5 (gfx1250) GNN pipeline: bf16 WMMA GEMMs, double-buffered TDM
// (tensor_load_to_lds) tile staging, segment-softmax attention. Wave32.
// ---------------------------------------------------------------------------

typedef __attribute__((ext_vector_type(16))) __bf16 v16bf;
typedef __attribute__((ext_vector_type(8)))  __bf16 v8bf;
typedef __attribute__((ext_vector_type(8)))  float  v8f;
typedef __attribute__((ext_vector_type(4)))  unsigned v4u;
typedef __attribute__((ext_vector_type(8)))  int      v8i;
typedef __attribute__((ext_vector_type(4)))  int      v4i;

#define N_NODES 10000
#define N_EDGES 160000
#define DIN 64
#define DE  16
#define H   256

#define BM 128
#define BN 128
#define BK 32
#define LPAD 8   // LDS row pad: 8 bf16 = 16B = 4 DWORDs

#define ACT_NONE 0
#define ACT_LRELU 1
#define ACT_ELU 2
#define ACT_RELU 3
#define ACT_SP 4
#define ACT_SIG 5

__device__ __forceinline__ float actf(float v, int act) {
  switch (act) {
    case ACT_LRELU: return v > 0.f ? v : 0.01f * v;
    case ACT_ELU:   return v > 0.f ? v : (__expf(v) - 1.f);
    case ACT_RELU:  return v > 0.f ? v : 0.f;
    case ACT_SP:    return (v > 20.f) ? v : __logf(1.f + __expf(v));
    case ACT_SIG:   return 1.f / (1.f + __expf(-v));
    default:        return v;
  }
}

__device__ __forceinline__ unsigned fenc(float f) {
  unsigned b = __float_as_uint(f);
  return (b & 0x80000000u) ? ~b : (b | 0x80000000u);
}
__device__ __forceinline__ float fdec(unsigned u) {
  unsigned b = (u & 0x80000000u) ? (u ^ 0x80000000u) : ~u;
  return __uint_as_float(b);
}

// ---------------------------------------------------------------------------
// TDM: 2D tile load global -> LDS, bf16 elements.
// D# per CDNA5 ISA ch.7/8: group0 = {count, lds_addr, global_addr, type=2},
// group1 = {data_size=2B, pad cfg, tensor_dim0/1, tile_dim0/1, dim0 stride}.
// LDS padding: 4 DWORDs (16B = LPAD bf16) after every 16 DWORDs (=BK bf16),
// matching the [rows][BK+LPAD] LDS arrays. OOB reads return zero -> M/K tails
// are zero-filled by hardware.
// 6-arg builtin form (clang-23): (v4u, v8i, v4i, v4i, v8i, i32 cpol).
// ---------------------------------------------------------------------------
__device__ __forceinline__ void tdm_load_2d(unsigned lds_off, const void* gptr,
                                            unsigned td0, unsigned td1,
                                            unsigned long long stride0,
                                            unsigned tile0, unsigned tile1) {
  unsigned long long ga = (unsigned long long)gptr;
  v4u g0;
  g0[0] = 1u;                                            // count=1, user mode
  g0[1] = lds_off;                                       // lds_addr (bytes)
  g0[2] = (unsigned)(ga & 0xffffffffu);                  // global_addr[31:0]
  g0[3] = (unsigned)((ga >> 32) & 0x01ffffffu) | (2u << 30);  // ga[56:32] | type=2
  unsigned w0 = (1u << 16)      // data_size: 2 bytes
              | (1u << 20)      // pad_enable
              | (3u << 22)      // pad_interval: 16 DWORDs
              | (3u << 25);     // pad_amount: 4 DWORDs
  v8i g1;
  g1[0] = (int)w0;
  g1[1] = (int)((td0 & 0xffffu) << 16);                  // abar=0 | tensor_dim0 lo
  g1[2] = (int)((td0 >> 16) | ((td1 & 0xffffu) << 16));  // td0 hi | td1 lo
  g1[3] = (int)((td1 >> 16) | (tile0 << 16));            // td1 hi | tile_dim0
  g1[4] = (int)(tile1 & 0xffffu);                        // tile_dim1 | tile_dim2=0
  g1[5] = (int)(unsigned)(stride0 & 0xffffffffu);        // dim0 stride lo
  g1[6] = (int)(unsigned)((stride0 >> 32) & 0xffffu);    // dim0 stride hi | dim1 stride=0
  g1[7] = 0;
  v4i z4; z4[0] = 0; z4[1] = 0; z4[2] = 0; z4[3] = 0;
  v8i z8;
  #pragma unroll
  for (int j = 0; j < 8; ++j) z8[j] = 0;
  __builtin_amdgcn_tensor_load_to_lds(g0, g1, z4, z4, z8, 0);
}

// ---------------------------------------------------------------------------
// Tiled bf16 WMMA GEMM:  C[M,Nc] = act( opA[M,K] @ W[Nc,K]^T + bias )
//   AMODE 0: A plain bf16 [M,lda], staged by TDM (2 TDM ops / stage)
//   AMODE 2: row m of A = concat( Xg[gidx[m], 0:K1], Eg[m, 0:K-K1] ),
//            A cooperatively staged, W by TDM (1 TDM op / stage)
//   EMODE 0: store fp32 C and/or bf16 Cbf
//   EMODE 1: atomicAdd( C[scat[m]*ldc + n], acc * scale[m] )
// 256 threads = 8 waves; wave owns a 16-row strip and 8 N-tiles of 16.
// Double-buffered: tile i+1's TDM is issued before tile i's compute; the
// per-wave in-order TENSORcnt lets us wait "<= nT outstanding" so only the
// current tile's DMA is forced complete.
// ---------------------------------------------------------------------------
template <int AMODE, int EMODE>
__global__ __launch_bounds__(256)
void k_gemm(const __bf16* __restrict__ A, int lda,
            const __bf16* __restrict__ Xg, int ldx, int K1,
            const __bf16* __restrict__ Eg, int lde,
            const int* __restrict__ gidx,
            const __bf16* __restrict__ W,
            const float* __restrict__ bias,
            int M, int Nc, int K,
            float* __restrict__ C, int ldc, int act,
            __bf16* __restrict__ Cbf,
            const int* __restrict__ scat, const float* __restrict__ scale)
{
  __shared__ __bf16 lA[2][BM][BK + LPAD];   // A frag: k = 8g+s (s<8), 16+8g+s-8 (s>=8)
  __shared__ __bf16 lB[2][BN][BK + LPAD];   // lB[b][n][k] = W[n0+n, k0+k]

  const int tid  = threadIdx.x;
  const int wave = tid >> 5;
  const int lane = tid & 31;
  const int g    = lane >> 4;
  const int l16  = lane & 15;
  const int m0   = blockIdx.x * BM;
  const int n0   = blockIdx.y * BN;

  v8f acc[8];
  #pragma unroll
  for (int t = 0; t < 8; ++t)
    #pragma unroll
    for (int j = 0; j < 8; ++j) acc[t][j] = 0.f;

  // ---- stage helper (TDM part, wave 0 only) ----
  auto stage_tdm = [&](int bf, int k0) {
    tdm_load_2d((unsigned)(unsigned long long)&lB[bf][0][0],
                &W[(size_t)n0 * K + k0],
                (unsigned)(K - k0), (unsigned)(Nc - n0),
                (unsigned long long)K, BK, BN);
    if (AMODE == 0)
      tdm_load_2d((unsigned)(unsigned long long)&lA[bf][0][0],
                  &A[(size_t)m0 * lda + k0],
                  (unsigned)(K - k0), (unsigned)(M - m0),
                  (unsigned long long)lda, BK, BM);
  };
  // ---- stage helper (cooperative gather+concat A, AMODE 2) ----
  auto stage_coop = [&](int bf, int k0) {
    for (int e = tid; e < BM * 4; e += 256) {
      int r = e >> 2, ck = (e & 3) * 8;
      int gm = m0 + r, gk = k0 + ck;
      v8bf v;
      #pragma unroll
      for (int j = 0; j < 8; ++j) v[j] = (__bf16)0.0f;
      if (gm < M && gk + 8 <= K) {
        if (gk + 8 <= K1) v = *(const v8bf*)&Xg[(size_t)gidx[gm] * ldx + gk];
        else              v = *(const v8bf*)&Eg[(size_t)gm * lde + (gk - K1)];
      }
      *(v8bf*)&lA[bf][r][ck] = v;
    }
  };

  // ---- prologue: stage tile 0 into buffer 0 ----
  if (wave == 0) stage_tdm(0, 0);
  if (AMODE == 2) stage_coop(0, 0);

  int buf = 0;
  for (int k0 = 0; k0 < K; k0 += BK) {
    const bool has_next = (k0 + BK < K);
    // issue next tile's DMA into the other buffer (overlaps this tile's math)
    if (has_next) {
      if (wave == 0) stage_tdm(buf ^ 1, k0 + BK);
      if (AMODE == 2) stage_coop(buf ^ 1, k0 + BK);
      if (tid == 0) __builtin_prefetch((const void*)&W[(size_t)n0 * K + k0 + BK], 0, 1);
    }
    // force only the CURRENT tile's TDM complete (next tile may stay in flight)
    if (wave == 0) {
      if (has_next) __builtin_amdgcn_s_wait_tensorcnt(AMODE == 0 ? 2 : 1);
      else          __builtin_amdgcn_s_wait_tensorcnt(0);
    }
    __syncthreads();

    // ---- A fragment: two contiguous 16B chunks per lane ----
    union { v16bf v; v8bf h[2]; } ua;
    const int ar = (wave << 4) + l16;
    ua.h[0] = *(const v8bf*)&lA[buf][ar][g * 8];
    ua.h[1] = *(const v8bf*)&lA[buf][ar][16 + g * 8];
    const v16bf aF = ua.v;

    #pragma unroll
    for (int t = 0; t < 8; ++t) {
      union { v16bf v; v8bf h[2]; } ub;
      const int br = t * 16 + l16;
      ub.h[0] = *(const v8bf*)&lB[buf][br][g * 8];
      ub.h[1] = *(const v8bf*)&lB[buf][br][16 + g * 8];
      acc[t] = __builtin_amdgcn_wmma_f32_16x16x32_bf16(
          false, aF, false, ub.v, (short)0, acc[t], false, false);
    }
    __syncthreads();
    buf ^= 1;
  }

  // ---- epilogue: C/D layout: VGPR v -> M = v + g*8 ; N = l16 ----
  #pragma unroll
  for (int t = 0; t < 8; ++t) {
    #pragma unroll
    for (int v = 0; v < 8; ++v) {
      int row = m0 + (wave << 4) + (g << 3) + v;
      int col = n0 + t * 16 + l16;
      if (row < M && col < Nc) {
        float val = acc[t][v];
        if (EMODE == 1) {
          atomicAdd(&C[(size_t)scat[row] * ldc + col], val * scale[row]);
        } else {
          if (bias) val += bias[col];
          val = actf(val, act);
          if (C)   C[(size_t)row * ldc + col]   = val;
          if (Cbf) Cbf[(size_t)row * ldc + col] = (__bf16)val;
        }
      }
    }
  }
}

// ---------------------------------------------------------------------------
// Small kernels
// ---------------------------------------------------------------------------
__global__ void k_cvt(const float* __restrict__ s, __bf16* __restrict__ d, int n) {
  int i = blockIdx.x * blockDim.x + threadIdx.x;
  if (i < n) d[i] = (__bf16)s[i];
}
__global__ void k_fill_f32(float* p, float v, int n) {
  int i = blockIdx.x * blockDim.x + threadIdx.x;
  if (i < n) p[i] = v;
}
__global__ void k_fill_u32(unsigned* p, unsigned v, int n) {
  int i = blockIdx.x * blockDim.x + threadIdx.x;
  if (i < n) p[i] = v;
}
__global__ void k_rowdot(const float* __restrict__ X, const float* __restrict__ w,
                         float* __restrict__ out, int n, int ld) {
  int i = blockIdx.x * blockDim.x + threadIdx.x;
  if (i >= n) return;
  const float* r = X + (size_t)i * ld;
  float s = 0.f;
  for (int k = 0; k < ld; ++k) s += r[k] * w[k];
  out[i] = s;
}
__global__ void k_rowdot_bf(const __bf16* __restrict__ X, const float* __restrict__ w,
                            float* __restrict__ out, int n, int ld) {
  int i = blockIdx.x * blockDim.x + threadIdx.x;
  if (i >= n) return;
  const __bf16* r = X + (size_t)i * ld;
  float s = 0.f;
  for (int k = 0; k < ld; ++k) s += (float)r[k] * w[k];
  out[i] = s;
}
__global__ void k_alpha_conv(const float* __restrict__ hl, const float* __restrict__ rdst,
                             const int* __restrict__ dstI, float* __restrict__ alpha,
                             unsigned* __restrict__ segmax, int E) {
  int e = blockIdx.x * blockDim.x + threadIdx.x;
  if (e >= E) return;
  int d = dstI[e];
  float a = hl[e] + rdst[d];
  a = a > 0.f ? a : 0.01f * a;
  alpha[e] = a;
  atomicMax(&segmax[d], fenc(a));
}
__global__ void k_alpha_gat(const float* __restrict__ asrc, const float* __restrict__ adst,
                            const int* __restrict__ srcI, const int* __restrict__ dstI,
                            float* __restrict__ alpha, unsigned* __restrict__ segmax, int E) {
  int e = blockIdx.x * blockDim.x + threadIdx.x;
  if (e >= E) return;
  int d = dstI[e];
  float a = asrc[srcI[e]] + adst[d];
  a = a > 0.f ? a : 0.01f * a;
  alpha[e] = a;
  atomicMax(&segmax[d], fenc(a));
}
__global__ void k_exp_sum(float* __restrict__ alpha, const unsigned* __restrict__ segmax,
                          const int* __restrict__ dstI, float* __restrict__ segsum, int E) {
  int e = blockIdx.x * blockDim.x + threadIdx.x;
  if (e >= E) return;
  int d = dstI[e];
  float v = __expf(alpha[e] - fdec(segmax[d]));
  alpha[e] = v;
  atomicAdd(&segsum[d], v);
}
__global__ void k_norm_alpha(float* __restrict__ alpha, const float* __restrict__ segsum,
                             const int* __restrict__ dstI, int E) {
  int e = blockIdx.x * blockDim.x + threadIdx.x;
  if (e >= E) return;
  alpha[e] = alpha[e] / (segsum[dstI[e]] + 1e-16f);
}
__global__ void k_gat_scatter(const float* __restrict__ xl, const int* __restrict__ srcI,
                              const int* __restrict__ dstI, const float* __restrict__ alpha,
                              float* __restrict__ acc, long long total) {
  long long idx = (long long)blockIdx.x * blockDim.x + threadIdx.x;
  if (idx >= total) return;
  int e = (int)(idx >> 8);
  int c = (int)(idx & 255);
  atomicAdd(&acc[(size_t)dstI[e] * H + c], xl[(size_t)srcI[e] * H + c] * alpha[e]);
}
__global__ void k_bias_elu(const float* __restrict__ acc, const float* __restrict__ bias,
                           float* __restrict__ hf, __bf16* __restrict__ hb, int total) {
  int idx = blockIdx.x * blockDim.x + threadIdx.x;
  if (idx >= total) return;
  int c = idx & 255;
  float v = acc[idx] + bias[c];
  v = v > 0.f ? v : (__expf(v) - 1.f);
  hf[idx] = v;
  hb[idx] = (__bf16)v;
}
__global__ void k_gru(const float* __restrict__ gi, const float* __restrict__ gh,
                      float* __restrict__ x, __bf16* __restrict__ xb, int total) {
  int idx = blockIdx.x * blockDim.x + threadIdx.x;
  if (idx >= total) return;
  int i = idx >> 8, c = idx & 255;
  size_t b = (size_t)i * (3 * H);
  float ir = gi[b + c], iz = gi[b + H + c], in = gi[b + 2 * H + c];
  float hr = gh[b + c], hz = gh[b + H + c], hn = gh[b + 2 * H + c];
  float r = 1.f / (1.f + __expf(-(ir + hr)));
  float z = 1.f / (1.f + __expf(-(iz + hz)));
  float n = tanhf(in + r * hn);
  float hid = x[idx];
  float o = (1.f - z) * n + z * hid;
  o = o > 0.f ? o : 0.f;
  x[idx] = o;
  xb[idx] = (__bf16)o;
}
__global__ void k_layernorm(const float* __restrict__ x, float* __restrict__ of,
                            __bf16* __restrict__ ob, int N) {
  __shared__ float red[H];
  int i = blockIdx.x, c = threadIdx.x;
  float v = x[(size_t)i * H + c];
  red[c] = v; __syncthreads();
  for (int s = H / 2; s > 0; s >>= 1) { if (c < s) red[c] += red[c + s]; __syncthreads(); }
  float mu = red[0] / (float)H; __syncthreads();
  float d = v - mu;
  red[c] = d * d; __syncthreads();
  for (int s = H / 2; s > 0; s >>= 1) { if (c < s) red[c] += red[c + s]; __syncthreads(); }
  float rstd = rsqrtf(red[0] / (float)H + 1e-5f);
  float o = d * rstd;
  of[(size_t)i * H + c] = o;
  ob[(size_t)i * H + c] = (__bf16)o;
}
__global__ void k_vec_head(const float* __restrict__ x, const float* __restrict__ w,
                           const float* __restrict__ b, float* __restrict__ out, int N) {
  int i = blockIdx.x * blockDim.x + threadIdx.x;
  if (i >= N) return;
  const float* r = x + (size_t)i * H;
  float s = b[0];
  for (int k = 0; k < H; ++k) s += r[k] * w[k];
  out[i] = 1.f / (1.f + __expf(-s));
}

// ---------------------------------------------------------------------------
// Host orchestration
// ---------------------------------------------------------------------------
static inline int ceil_div(int a, int b) { return (a + b - 1) / b; }

static void gemm(hipStream_t s,
                 const __bf16* A, int lda,
                 const __bf16* X, int ldx, int K1,
                 const __bf16* Eg, int lde,
                 const int* gidx, int amode,
                 const __bf16* W, const float* bias,
                 int M, int Nc, int K,
                 float* C, int ldc, int act, __bf16* Cbf,
                 const int* scat, const float* scale, int emode) {
  dim3 gr(ceil_div(M, BM), ceil_div(Nc, BN));
  if (amode == 2)
    k_gemm<2, 0><<<gr, 256, 0, s>>>(A, lda, X, ldx, K1, Eg, lde, gidx,
                                    W, bias, M, Nc, K, C, ldc, act, Cbf, scat, scale);
  else if (emode == 1)
    k_gemm<0, 1><<<gr, 256, 0, s>>>(A, lda, X, ldx, K1, Eg, lde, gidx,
                                    W, bias, M, Nc, K, C, ldc, act, Cbf, scat, scale);
  else
    k_gemm<0, 0><<<gr, 256, 0, s>>>(A, lda, X, ldx, K1, Eg, lde, gidx,
                                    W, bias, M, Nc, K, C, ldc, act, Cbf, scat, scale);
}
static void cvt(hipStream_t s, const float* src, __bf16* dst, int n) {
  k_cvt<<<ceil_div(n, 256), 256, 0, s>>>(src, dst, n);
}

struct BB {
  const float *enter_w, *enter_b, *w1, *w2, *att_l, *att_r, *cbias;
  const float *gat_w[3], *gat_as[3], *gat_ad[3], *gat_b[3];
  const float *gih[4], *ghh[4], *bih[4], *bhh[4];
};
static BB make_bb(void* const* d_in, int base) {
  BB b;
  auto F = [&](int i) { return (const float*)d_in[base + i]; };
  b.enter_w = F(0); b.enter_b = F(1); b.w1 = F(2); b.w2 = F(3);
  b.att_l = F(4); b.att_r = F(5); b.cbias = F(6);
  for (int g = 0; g < 3; ++g) {
    int o = 7 + g * 4;
    b.gat_w[g] = F(o); b.gat_as[g] = F(o + 1); b.gat_ad[g] = F(o + 2); b.gat_b[g] = F(o + 3);
  }
  for (int i = 0; i < 4; ++i) {
    int o = 19 + i * 4;
    b.gih[i] = F(o); b.ghh[i] = F(o + 1); b.bih[i] = F(o + 2); b.bhh[i] = F(o + 3);
  }
  return b;
}

struct WS {
  __bf16 *x_in_bf, *ea_bf;
  __bf16 *wb_enter, *wb_w1, *wb_w2, *wb_gat[3], *wb_gih[4], *wb_ghh[4], *wb_head;
  float *xf, *hf, *xlf, *gi, *gh, *acc, *lnf;
  __bf16 *xb, *hb, *he, *lnb, *tb0, *tb1;
  float *hl, *alpha, *rdst, *asrc, *adst, *segsum;
  unsigned* segmax;
};

static void run_backbone(hipStream_t s, const BB& p, const WS& w,
                         const int* srcI, const int* dstI) {
  const int NH = N_NODES * H;
  cvt(s, p.enter_w, w.wb_enter, H * DIN);
  cvt(s, p.w1, w.wb_w1, H * (H + DE));
  cvt(s, p.w2, w.wb_w2, H * H);
  for (int g = 0; g < 3; ++g) cvt(s, p.gat_w[g], w.wb_gat[g], H * H);
  for (int i = 0; i < 4; ++i) { cvt(s, p.gih[i], w.wb_gih[i], 3 * H * H);
                                cvt(s, p.ghh[i], w.wb_ghh[i], 3 * H * H); }

  // x = leaky_relu(x_in @ enter_w.T + b)
  gemm(s, w.x_in_bf, DIN, nullptr, 0, 0, nullptr, 0, nullptr, 0,
       w.wb_enter, p.enter_b, N_NODES, H, DIN,
       w.xf, H, ACT_LRELU, w.xb, nullptr, nullptr, 0);

  // ---- Edge2DConv ----
  gemm(s, nullptr, 0, w.xb, H, H, w.ea_bf, DE, srcI, 2,
       w.wb_w1, nullptr, N_EDGES, H, H + DE,
       nullptr, H, ACT_LRELU, w.he, nullptr, nullptr, 0);
  k_rowdot_bf<<<ceil_div(N_EDGES, 256), 256, 0, s>>>(w.he, p.att_l, w.hl, N_EDGES, H);
  k_rowdot<<<ceil_div(N_NODES, 256), 256, 0, s>>>(w.xf, p.att_r, w.rdst, N_NODES, H);
  k_fill_u32<<<ceil_div(N_NODES, 256), 256, 0, s>>>(w.segmax, 0u, N_NODES);
  k_fill_f32<<<ceil_div(N_NODES, 256), 256, 0, s>>>(w.segsum, 0.f, N_NODES);
  k_alpha_conv<<<ceil_div(N_EDGES, 256), 256, 0, s>>>(w.hl, w.rdst, dstI, w.alpha, w.segmax, N_EDGES);
  k_exp_sum<<<ceil_div(N_EDGES, 256), 256, 0, s>>>(w.alpha, w.segmax, dstI, w.segsum, N_EDGES);
  k_norm_alpha<<<ceil_div(N_EDGES, 256), 256, 0, s>>>(w.alpha, w.segsum, dstI, N_EDGES);
  k_fill_f32<<<ceil_div(NH, 256), 256, 0, s>>>(w.acc, 0.f, NH);
  gemm(s, w.he, H, nullptr, 0, 0, nullptr, 0, nullptr, 0,
       w.wb_w2, nullptr, N_EDGES, H, H,
       w.acc, H, ACT_NONE, nullptr, dstI, w.alpha, 1);
  k_bias_elu<<<ceil_div(NH, 256), 256, 0, s>>>(w.acc, p.cbias, w.hf, w.hb, NH);
  gemm(s, w.hb, H, nullptr, 0, 0, nullptr, 0, nullptr, 0,
       w.wb_gih[0], p.bih[0], N_NODES, 3 * H, H, w.gi, 3 * H, ACT_NONE, nullptr, nullptr, nullptr, 0);
  gemm(s, w.xb, H, nullptr, 0, 0, nullptr, 0, nullptr, 0,
       w.wb_ghh[0], p.bhh[0], N_NODES, 3 * H, H, w.gh, 3 * H, ACT_NONE, nullptr, nullptr, nullptr, 0);
  k_gru<<<ceil_div(NH, 256), 256, 0, s>>>(w.gi, w.gh, w.xf, w.xb, NH);

  // ---- 3 x GATConv + GRU ----
  for (int g = 0; g < 3; ++g) {
    gemm(s, w.xb, H, nullptr, 0, 0, nullptr, 0, nullptr, 0,
         w.wb_gat[g], nullptr, N_NODES, H, H, w.xlf, H, ACT_NONE, nullptr, nullptr, nullptr, 0);
    k_rowdot<<<ceil_div(N_NODES, 256), 256, 0, s>>>(w.xlf, p.gat_as[g], w.asrc, N_NODES, H);
    k_rowdot<<<ceil_div(N_NODES, 256), 256, 0, s>>>(w.xlf, p.gat_ad[g], w.adst, N_NODES, H);
    k_fill_u32<<<ceil_div(N_NODES, 256), 256, 0, s>>>(w.segmax, 0u, N_NODES);
    k_fill_f32<<<ceil_div(N_NODES, 256), 256, 0, s>>>(w.segsum, 0.f, N_NODES);
    k_alpha_gat<<<ceil_div(N_EDGES, 256), 256, 0, s>>>(w.asrc, w.adst, srcI, dstI, w.alpha, w.segmax, N_EDGES);
    k_exp_sum<<<ceil_div(N_EDGES, 256), 256, 0, s>>>(w.alpha, w.segmax, dstI, w.segsum, N_EDGES);
    k_norm_alpha<<<ceil_div(N_EDGES, 256), 256, 0, s>>>(w.alpha, w.segsum, dstI, N_EDGES);
    k_fill_f32<<<ceil_div(NH, 256), 256, 0, s>>>(w.acc, 0.f, NH);
    long long tot = (long long)N_EDGES * H;
    k_gat_scatter<<<(unsigned)(tot / 256), 256, 0, s>>>(w.xlf, srcI, dstI, w.alpha, w.acc, tot);
    k_bias_elu<<<ceil_div(NH, 256), 256, 0, s>>>(w.acc, p.gat_b[g], w.hf, w.hb, NH);
    gemm(s, w.hb, H, nullptr, 0, 0, nullptr, 0, nullptr, 0,
         w.wb_gih[g + 1], p.bih[g + 1], N_NODES, 3 * H, H, w.gi, 3 * H, ACT_NONE, nullptr, nullptr, nullptr, 0);
    gemm(s, w.xb, H, nullptr, 0, 0, nullptr, 0, nullptr, 0,
         w.wb_ghh[g + 1], p.bhh[g + 1], N_NODES, 3 * H, H, w.gh, 3 * H, ACT_NONE, nullptr, nullptr, nullptr, 0);
    k_gru<<<ceil_div(NH, 256), 256, 0, s>>>(w.gi, w.gh, w.xf, w.xb, NH);
  }
  k_layernorm<<<N_NODES, H, 0, s>>>(w.xf, w.lnf, w.lnb, N_NODES);
}

static void run_head(hipStream_t s, const WS& w, void* const* d_in,
                     int first_leaf, int nlayers, float* out) {
  const __bf16* prev = w.lnb;
  for (int i = 0; i < nlayers; ++i) {
    const float* wt = (const float*)d_in[first_leaf + 2 * i];
    const float* bs = (const float*)d_in[first_leaf + 2 * i + 1];
    cvt(s, wt, w.wb_head, H * H);
    bool last = (i == nlayers - 1);
    __bf16* obf = last ? nullptr : ((i & 1) ? w.tb1 : w.tb0);
    gemm(s, prev, H, nullptr, 0, 0, nullptr, 0, nullptr, 0,
         w.wb_head, bs, N_NODES, H, H,
         last ? out : nullptr, H, last ? ACT_NONE : ACT_SP, obf,
         nullptr, nullptr, 0);
    prev = obf;
  }
}

extern "C" void kernel_launch(void* const* d_in, const int* in_sizes, int n_in,
                              void* d_out, int out_size, void* d_ws, size_t ws_size,
                              hipStream_t stream) {
  (void)in_sizes; (void)n_in; (void)out_size; (void)ws_size;

  const float* x_in = (const float*)d_in[0];
  const int* edge_index = (const int*)d_in[1];
  const int* srcI = edge_index;
  const int* dstI = edge_index + N_EDGES;
  const float* edge_attr = (const float*)d_in[2];

  BB bb_vec = make_bb(d_in, 4);
  BB bb_lnk = make_bb(d_in, 41);
  BB bb_bnd = make_bb(d_in, 88);

  char* base = (char*)d_ws;
  size_t off = 0;
  auto alloc = [&](size_t bytes) -> void* {
    void* p = base + off;
    off = (off + bytes + 255) & ~(size_t)255;
    return p;
  };
  WS w;
  w.x_in_bf = (__bf16*)alloc((size_t)N_NODES * DIN * 2);
  w.ea_bf   = (__bf16*)alloc((size_t)N_EDGES * DE * 2);
  w.wb_enter = (__bf16*)alloc((size_t)H * DIN * 2);
  w.wb_w1    = (__bf16*)alloc((size_t)H * (H + DE) * 2);
  w.wb_w2    = (__bf16*)alloc((size_t)H * H * 2);
  for (int g = 0; g < 3; ++g) w.wb_gat[g] = (__bf16*)alloc((size_t)H * H * 2);
  for (int i = 0; i < 4; ++i) { w.wb_gih[i] = (__bf16*)alloc((size_t)3 * H * H * 2);
                                w.wb_ghh[i] = (__bf16*)alloc((size_t)3 * H * H * 2); }
  w.wb_head = (__bf16*)alloc((size_t)H * H * 2);
  w.xf  = (float*)alloc((size_t)N_NODES * H * 4);
  w.hf  = (float*)alloc((size_t)N_NODES * H * 4);
  w.xlf = (float*)alloc((size_t)N_NODES * H * 4);
  w.gi  = (float*)alloc((size_t)N_NODES * 3 * H * 4);
  w.gh  = (float*)alloc((size_t)N_NODES * 3 * H * 4);
  w.acc = (float*)alloc((size_t)N_NODES * H * 4);
  w.lnf = (float*)alloc((size_t)N_NODES * H * 4);
  w.xb  = (__bf16*)alloc((size_t)N_NODES * H * 2);
  w.hb  = (__bf16*)alloc((size_t)N_NODES * H * 2);
  w.lnb = (__bf16*)alloc((size_t)N_NODES * H * 2);
  w.tb0 = (__bf16*)alloc((size_t)N_NODES * H * 2);
  w.tb1 = (__bf16*)alloc((size_t)N_NODES * H * 2);
  w.he  = (__bf16*)alloc((size_t)N_EDGES * H * 2);
  w.hl    = (float*)alloc((size_t)N_EDGES * 4);
  w.alpha = (float*)alloc((size_t)N_EDGES * 4);
  w.rdst  = (float*)alloc((size_t)N_NODES * 4);
  w.asrc  = (float*)alloc((size_t)N_NODES * 4);
  w.adst  = (float*)alloc((size_t)N_NODES * 4);
  w.segsum = (float*)alloc((size_t)N_NODES * 4);
  w.segmax = (unsigned*)alloc((size_t)N_NODES * 4);

  cvt(stream, x_in, w.x_in_bf, N_NODES * DIN);
  cvt(stream, edge_attr, w.ea_bf, N_EDGES * DE);

  float* out = (float*)d_out;
  const size_t OUT_VEC = 0;
  const size_t OUT_XA  = N_NODES;
  const size_t OUT_XB  = OUT_XA + (size_t)N_NODES * H;
  const size_t OUT_XBM = OUT_XB + (size_t)N_NODES * H;

  run_backbone(stream, bb_vec, w, srcI, dstI);
  k_vec_head<<<ceil_div(N_NODES, 256), 256, 0, stream>>>(
      w.lnf, (const float*)d_in[39], (const float*)d_in[40], out + OUT_VEC, N_NODES);

  run_backbone(stream, bb_lnk, w, srcI, dstI);
  run_head(stream, w, d_in, 76, 3, out + OUT_XA);
  run_head(stream, w, d_in, 82, 3, out + OUT_XB);

  run_backbone(stream, bb_bnd, w, srcI, dstI);
  run_head(stream, w, d_in, 123, 2, out + OUT_XBM);
}
